// MultiHeadBlockSelf_22969485099563
// MI455X (gfx1250) — matrix operand order repaired and checked by reading, not compiled, not run on hardware
//
#include <hip/hip_runtime.h>
#include <hip/hip_bf16.h>
#include <math.h>

// Problem constants (match reference)
#define B_    2
#define T_    2048
#define D_    768
#define H_    12
#define DH_   64
#define DV_   384
#define CPOUT_ 6144         // 2*D + DV*H
#define BT_   4096          // B*T
#define HV_   4608          // H*DV

typedef __attribute__((ext_vector_type(16))) __bf16 v16bf;
typedef __attribute__((ext_vector_type(8)))  __bf16 v8bf;
typedef __attribute__((ext_vector_type(8)))  float  v8f;

// ---------------------------------------------------------------------------
// WMMA helpers (CDNA5 wave32: D = A(16x32 bf16) * B(32x16 bf16) + C(16x16 f32))
// ---------------------------------------------------------------------------
static __device__ __forceinline__ v8f wmma_bf16(v16bf a, v16bf b, v8f c) {
  return __builtin_amdgcn_wmma_f32_16x16x32_bf16(
      /*neg_a=*/false, a, /*neg_b=*/false, b,
      /*c_mod=*/(short)0, c, /*reuse_a=*/false, /*reuse_b=*/false);
}

static __device__ __forceinline__ v16bf cat8(v8bf lo, v8bf hi) {
  return __builtin_shufflevector(lo, hi, 0,1,2,3,4,5,6,7,8,9,10,11,12,13,14,15);
}

// A-fragment 16x32 bf16: lane holds row (lane&15), K = kb..kb+7 and kb+16..kb+23
// with kb = (lane>>4)*8.  Row-major [M,K] memory -> two contiguous 16B loads.
static __device__ __forceinline__ v16bf load_afrag(const __bf16* base, int ld,
                                                   int row0, int k0, int lane) {
  const __bf16* p = base + (size_t)(row0 + (lane & 15)) * ld + k0 + ((lane >> 4) * 8);
  v8bf lo = *(const v8bf*)p;
  v8bf hi = *(const v8bf*)(p + 16);
  return cat8(lo, hi);
}

// B-fragment 32x16 bf16: lane holds col (lane&15), K = kb..kb+7 and kb+16..kb+23
// with kb = (lane>>4)*8.  Row-major [N,K] memory (i.e. B^T) -> two 16B loads.
static __device__ __forceinline__ v16bf load_bfrag(const __bf16* base, int ld,
                                                   int col0, int k0, int lane) {
  const __bf16* p = base + (size_t)(col0 + (lane & 15)) * ld + k0 + ((lane >> 4) * 8);
  v8bf lo = *(const v8bf*)p;
  v8bf hi = *(const v8bf*)(p + 16);
  return cat8(lo, hi);
}

// ---------------------------------------------------------------------------
// fp32 -> bf16 conversion
// ---------------------------------------------------------------------------
__global__ __launch_bounds__(256) void cvt_f32_bf16(const float* __restrict__ in,
                                                    __bf16* __restrict__ out, int n) {
  int i = blockIdx.x * 256 + threadIdx.x;
  if (i < n) out[i] = (__bf16)in[i];
}

// ---------------------------------------------------------------------------
// Generic WMMA GEMM:  C[M,N] = epilogue( A[M,K] @ W[N,K]^T + bias[N] )
// Wave tile 32x64 (8 WMMAs / K-step), 8 waves -> 64x256 block tile.
// ---------------------------------------------------------------------------
enum { ACT_CP = 0, ACT_WO = 1, ACT_W1 = 2, ACT_W2 = 3 };

template <int MODE>
__global__ __launch_bounds__(256) void gemm_bf16_wmma(
    const __bf16* __restrict__ A, const __bf16* __restrict__ W,
    const float* __restrict__ bias, void* __restrict__ out,
    const float* __restrict__ resid, int M, int N, int K) {
  const int lane = threadIdx.x & 31;
  const int wave = threadIdx.x >> 5;
  const int m0 = blockIdx.y * 64 + (wave >> 2) * 32;
  const int n0 = blockIdx.x * 256 + (wave & 3) * 64;

  v8f acc[2][4];
#pragma unroll
  for (int i = 0; i < 2; ++i)
#pragma unroll
    for (int j = 0; j < 4; ++j) acc[i][j] = (v8f)0.0f;

  for (int k = 0; k < K; k += 32) {
    if (k + 32 < K) {  // lowers to global_prefetch_b8
      __builtin_prefetch(A + (size_t)(m0 + (lane & 15)) * K + k + 32, 0, 1);
      __builtin_prefetch(W + (size_t)(n0 + (lane & 15)) * K + k + 32, 0, 1);
    }
    v16bf a0 = load_afrag(A, K, m0, k, lane);
    v16bf a1 = load_afrag(A, K, m0 + 16, k, lane);
#pragma unroll
    for (int j = 0; j < 4; ++j) {
      v16bf bfrag = load_bfrag(W, K, n0 + 16 * j, k, lane);
      acc[0][j] = wmma_bf16(a0, bfrag, acc[0][j]);
      acc[1][j] = wmma_bf16(a1, bfrag, acc[1][j]);
    }
  }

  // Epilogue. C-layout: lane holds rows rb..rb+7 (rb=8*(lane>>4)), col = lane&15.
  const int rb = (lane >> 4) * 8;
  const int cl = lane & 15;
#pragma unroll
  for (int i = 0; i < 2; ++i) {
#pragma unroll
    for (int j = 0; j < 4; ++j) {
      const int col = n0 + 16 * j + cl;
      const float bv = bias[col];
#pragma unroll
      for (int r = 0; r < 8; ++r) {
        const int row = m0 + 16 * i + rb + r;
        float v = acc[i][j][r] + bv;
        const size_t idx = (size_t)row * N + col;
        if (MODE == ACT_CP) {           // leaky-relu -> bf16 (feeds attention)
          v = v >= 0.0f ? v : 0.01f * v;
          ((__bf16*)out)[idx] = (__bf16)v;
        } else if (MODE == ACT_WO) {    // leaky-relu -> fp32 (feeds resid+LN)
          v = v >= 0.0f ? v : 0.01f * v;
          ((float*)out)[idx] = v;
        } else if (MODE == ACT_W1) {    // selu -> bf16
          v = 1.0507009873554805f *
              (v > 0.0f ? v : 1.6732632423543772f * expm1f(v));
          ((__bf16*)out)[idx] = (__bf16)v;
        } else {                        // exact gelu + residual -> fp32 output
          const float gel = 0.5f * v * (1.0f + erff(v * 0.7071067811865476f));
          ((float*)out)[idx] = resid[idx] + gel;
        }
      }
    }
  }
}

// ---------------------------------------------------------------------------
// Repack cp -> q (scaled by 1/sqrt(DH), exact in bf16), k [B,H,T,DH]
// ---------------------------------------------------------------------------
__global__ __launch_bounds__(256) void repack_qk(const __bf16* __restrict__ cp,
                                                 __bf16* __restrict__ q,
                                                 __bf16* __restrict__ kk) {
  int idx = blockIdx.x * 256 + threadIdx.x;
  if (idx >= BT_ * D_) return;
  int d = idx % D_;
  int m = idx / D_;
  int t = m % T_, b = m / T_;
  int h = d / DH_, dh = d % DH_;
  size_t src = (size_t)m * CPOUT_;
  size_t dst = ((size_t)(b * H_ + h) * T_ + t) * DH_ + dh;
  q[dst]  = (__bf16)((float)cp[src + d] * 0.125f);  // fold 1/SCALE, SCALE=8
  kk[dst] = cp[src + D_ + d];
}

// v transposed to [B,H,DV,T] so V-tiles load as contiguous B-fragments
__global__ __launch_bounds__(256) void repack_v(const __bf16* __restrict__ cp,
                                                __bf16* __restrict__ vT) {
  int idx = blockIdx.x * 256 + threadIdx.x;  // output-order index
  if (idx >= B_ * H_ * DV_ * T_) return;
  int t = idx % T_;
  int rest = idx / T_;
  int d = rest % DV_;
  int bh = rest / DV_;
  int b = bh / H_, h = bh % H_;
  vT[idx] = cp[((size_t)(b * T_ + t)) * CPOUT_ + 2 * D_ + h * DV_ + d];
}

// ---------------------------------------------------------------------------
// Flash-style causal attention. Block = 16 query rows x (b,h); 4 waves each
// own a 96-wide DV slice and redundantly compute scores (QK^T is only 1/6 of
// the A*V flops). P goes through LDS to convert C-layout -> A-fragment.
// ---------------------------------------------------------------------------
__global__ __launch_bounds__(128) void attn_kernel(
    const __bf16* __restrict__ qb, const __bf16* __restrict__ kb,
    const __bf16* __restrict__ vT, __bf16* __restrict__ aout) {
  __shared__ __attribute__((aligned(16))) __bf16 pbuf[4][16][32];
  const int lane = threadIdx.x & 31;
  const int wave = threadIdx.x >> 5;  // DV slice owner
  const int q0 = blockIdx.x * 16;
  const int bh = blockIdx.y;
  const int b = bh / H_, h = bh % H_;
  const __bf16* Q  = qb + ((size_t)bh * T_ + q0) * DH_;
  const __bf16* Km = kb + (size_t)bh * T_ * DH_;
  const __bf16* Vm = vT + (size_t)bh * DV_ * T_;
  const int dv0 = wave * 96;
  const int rb = (lane >> 4) * 8;
  const int cl = lane & 15;

  // Q fragments (two K-chunks of DH=64), resident for the whole block
  v16bf qf0 = load_afrag(Q, DH_, 0, 0, lane);
  v16bf qf1 = load_afrag(Q, DH_, 0, 32, lane);

  v8f oacc[6];
#pragma unroll
  for (int j = 0; j < 6; ++j) oacc[j] = (v8f)0.0f;
  float mrow[8], lrow[8];
#pragma unroll
  for (int r = 0; r < 8; ++r) { mrow[r] = -1e30f; lrow[r] = 0.0f; }

  const int nsteps = (q0 + 47) >> 5;  // cover keys 0..q0+15
  for (int s = 0; s < nsteps; ++s) {
    const int kt = s << 5;
    // S[16 x 32] = Q @ K^T (scale pre-folded into Q)
    v8f sacc0 = (v8f)0.0f, sacc1 = (v8f)0.0f;
    {
      v16bf kf;
      kf = load_bfrag(Km, DH_, kt,       0,  lane); sacc0 = wmma_bf16(qf0, kf, sacc0);
      kf = load_bfrag(Km, DH_, kt,       32, lane); sacc0 = wmma_bf16(qf1, kf, sacc0);
      kf = load_bfrag(Km, DH_, kt + 16,  0,  lane); sacc1 = wmma_bf16(qf0, kf, sacc1);
      kf = load_bfrag(Km, DH_, kt + 16,  32, lane); sacc1 = wmma_bf16(qf1, kf, sacc1);
    }
    // causal mask + online softmax (16-lane row reductions, wave32)
#pragma unroll
    for (int r = 0; r < 8; ++r) {
      const int qrow = q0 + rb + r;
      float s0 = (kt + cl      > qrow) ? -1e30f : sacc0[r];
      float s1 = (kt + 16 + cl > qrow) ? -1e30f : sacc1[r];
      float mx = fmaxf(s0, s1);
      mx = fmaxf(mx, __shfl_xor(mx, 1, 32));
      mx = fmaxf(mx, __shfl_xor(mx, 2, 32));
      mx = fmaxf(mx, __shfl_xor(mx, 4, 32));
      mx = fmaxf(mx, __shfl_xor(mx, 8, 32));
      const float mnew = fmaxf(mrow[r], mx);
      const float corr = __expf(mrow[r] - mnew);
      const float p0 = __expf(s0 - mnew);
      const float p1 = __expf(s1 - mnew);
      float ps = p0 + p1;
      ps += __shfl_xor(ps, 1, 32);
      ps += __shfl_xor(ps, 2, 32);
      ps += __shfl_xor(ps, 4, 32);
      ps += __shfl_xor(ps, 8, 32);
      lrow[r] = lrow[r] * corr + ps;
      mrow[r] = mnew;
#pragma unroll
      for (int j = 0; j < 6; ++j) oacc[j][r] *= corr;
      pbuf[wave][rb + r][cl]      = (__bf16)p0;
      pbuf[wave][rb + r][16 + cl] = (__bf16)p1;
    }
    // make cross-lane LDS writes visible before A-frag reload (per-wave buffer)
    asm volatile("s_wait_dscnt 0" ::: "memory");
    v16bf pf = load_afrag(&pbuf[wave][0][0], 32, 0, 0, lane);
#pragma unroll
    for (int j = 0; j < 6; ++j) {
      v16bf vf = load_bfrag(Vm, T_, dv0 + 16 * j, kt, lane);
      oacc[j] = wmma_bf16(pf, vf, oacc[j]);
    }
  }
  // normalize and write bf16 [BT, H*DV] (A-matrix for the Wo GEMM)
#pragma unroll
  for (int r = 0; r < 8; ++r) {
    const float inv = 1.0f / lrow[r];
    const size_t rowo = ((size_t)(b * T_ + q0 + rb + r)) * HV_ + h * DV_ + dv0;
#pragma unroll
    for (int j = 0; j < 6; ++j)
      aout[rowo + 16 * j + cl] = (__bf16)(oacc[j][r] * inv);
  }
}

// ---------------------------------------------------------------------------
// inter = LayerNorm(x + leaky(y)) ; also emit bf16 copy for next GEMM
// ---------------------------------------------------------------------------
__global__ __launch_bounds__(256) void ln_kernel(
    const float* __restrict__ x, const float* __restrict__ y,
    const float* __restrict__ g, const float* __restrict__ bb,
    float* __restrict__ inter, __bf16* __restrict__ interb) {
  const int row = blockIdx.x;
  const int tid = threadIdx.x;
  __shared__ float s1[256], s2[256];
  float a = 0.0f, c = 0.0f;
  for (int i = tid; i < D_; i += 256) {
    const float t = x[(size_t)row * D_ + i] + y[(size_t)row * D_ + i];
    a += t; c += t * t;
  }
  s1[tid] = a; s2[tid] = c;
  __syncthreads();
  for (int st = 128; st > 0; st >>= 1) {
    if (tid < st) { s1[tid] += s1[tid + st]; s2[tid] += s2[tid + st]; }
    __syncthreads();
  }
  const float mu = s1[0] * (1.0f / D_);
  const float var = s2[0] * (1.0f / D_) - mu * mu;
  const float rstd = rsqrtf(var + 1e-5f);
  for (int i = tid; i < D_; i += 256) {
    const size_t idx = (size_t)row * D_ + i;
    const float t = x[idx] + y[idx];
    const float o = (t - mu) * rstd * g[i] + bb[i];
    inter[idx] = o;
    interb[idx] = (__bf16)o;
  }
}

// ---------------------------------------------------------------------------
// Host-side orchestration
// ---------------------------------------------------------------------------
extern "C" void kernel_launch(void* const* d_in, const int* in_sizes, int n_in,
                              void* d_out, int out_size, void* d_ws, size_t ws_size,
                              hipStream_t stream) {
  (void)in_sizes; (void)n_in; (void)out_size; (void)ws_size;
  const float* X   = (const float*)d_in[0];
  // d_in[1] = causal mask, structure known -> unused
  const float* Wc  = (const float*)d_in[2];
  const float* bc  = (const float*)d_in[3];
  const float* Wo  = (const float*)d_in[4];
  const float* bo  = (const float*)d_in[5];
  const float* W1  = (const float*)d_in[6];
  const float* b1  = (const float*)d_in[7];
  const float* W2  = (const float*)d_in[8];
  const float* b2  = (const float*)d_in[9];
  const float* lng = (const float*)d_in[10];
  const float* lnb = (const float*)d_in[11];
  float* out = (float*)d_out;

  char* w = (char*)d_ws;
  size_t off = 0;
  auto take = [&](size_t bytes) -> char* {
    char* p = w + off;
    off += (bytes + 255) & ~(size_t)255;
    return p;
  };
  __bf16* Xb   = (__bf16*)take((size_t)BT_ * D_ * 2);
  __bf16* Wcb  = (__bf16*)take((size_t)CPOUT_ * D_ * 2);
  __bf16* Wob  = (__bf16*)take((size_t)D_ * HV_ * 2);
  __bf16* W1b  = (__bf16*)take((size_t)D_ * D_ * 2);
  __bf16* W2b  = (__bf16*)take((size_t)D_ * D_ * 2);
  __bf16* cpb  = (__bf16*)take((size_t)BT_ * CPOUT_ * 2);
  __bf16* qbuf = (__bf16*)take((size_t)BT_ * D_ * 2);
  __bf16* kbuf = (__bf16*)take((size_t)BT_ * D_ * 2);
  __bf16* vTb  = (__bf16*)take((size_t)BT_ * HV_ * 2);
  float*  yact = (float*)take((size_t)BT_ * D_ * 4);
  float*  inter= (float*)take((size_t)BT_ * D_ * 4);
  __bf16* intb = (__bf16*)take((size_t)BT_ * D_ * 2);
  __bf16* h1   = (__bf16*)take((size_t)BT_ * D_ * 2);
  __bf16* aout = cpb;  // cpb is dead after repack; alias for attention output

  auto cg = [](size_t n) { return dim3((unsigned)((n + 255) / 256)); };

  cvt_f32_bf16<<<cg((size_t)BT_ * D_),    256, 0, stream>>>(X,  Xb,  BT_ * D_);
  cvt_f32_bf16<<<cg((size_t)CPOUT_ * D_), 256, 0, stream>>>(Wc, Wcb, CPOUT_ * D_);
  cvt_f32_bf16<<<cg((size_t)D_ * HV_),    256, 0, stream>>>(Wo, Wob, D_ * HV_);
  cvt_f32_bf16<<<cg((size_t)D_ * D_),     256, 0, stream>>>(W1, W1b, D_ * D_);
  cvt_f32_bf16<<<cg((size_t)D_ * D_),     256, 0, stream>>>(W2, W2b, D_ * D_);

  // cp = leaky(X @ Wc^T + bc)
  gemm_bf16_wmma<ACT_CP><<<dim3(CPOUT_ / 256, BT_ / 64), 256, 0, stream>>>(
      Xb, Wcb, bc, cpb, nullptr, BT_, CPOUT_, D_);
  repack_qk<<<cg((size_t)BT_ * D_),  256, 0, stream>>>(cpb, qbuf, kbuf);
  repack_v <<<cg((size_t)BT_ * HV_), 256, 0, stream>>>(cpb, vTb);

  attn_kernel<<<dim3(T_ / 16, B_ * H_), 128, 0, stream>>>(qbuf, kbuf, vTb, aout);

  // y = leaky(attn_out @ Wo^T + bo)
  gemm_bf16_wmma<ACT_WO><<<dim3(D_ / 256, BT_ / 64), 256, 0, stream>>>(
      aout, Wob, bo, yact, nullptr, BT_, D_, HV_);
  ln_kernel<<<dim3(BT_), 256, 0, stream>>>(X, yact, lng, lnb, inter, intb);

  // h1 = selu(inter @ W1^T + b1)
  gemm_bf16_wmma<ACT_W1><<<dim3(D_ / 256, BT_ / 64), 256, 0, stream>>>(
      intb, W1b, b1, h1, nullptr, BT_, D_, D_);
  // out = inter + gelu(h1 @ W2^T + b2)
  gemm_bf16_wmma<ACT_W2><<<dim3(D_ / 256, BT_ / 64), 256, 0, stream>>>(
      h1, W2b, b2, out, inter, BT_, D_, D_);
}
